// DeformBasicBlock_76922864271946
// MI455X (gfx1250) — matrix operand order repaired and checked by reading, not compiled, not run on hardware
//
#include <hip/hip_runtime.h>

// ---------------- problem constants ----------------
constexpr int kB  = 4;
constexpr int kC  = 256;
constexpr int kH  = 56;
constexpr int kW  = 56;
constexpr int kHW = kH * kW;          // 3136
constexpr int kM  = kB * kHW;         // 12544
constexpr int kK  = kC * 9;           // 2304
constexpr int kN  = kC;               // 256
constexpr int kOffC = 18;

typedef __attribute__((ext_vector_type(16))) __bf16 v16bf;
typedef __attribute__((ext_vector_type(8)))  float  v8f;
typedef __attribute__((ext_vector_type(4)))  unsigned int v4u;

union FragBF { v16bf v; v4u q[2]; };

// CDNA5 async global->LDS path (ASYNCcnt); fall back to register staging if absent.
#if defined(__AMDGCN__) && __has_builtin(__builtin_amdgcn_global_load_async_to_lds_b128)
#define USE_ASYNC_LDS 1
#else
#define USE_ASYNC_LDS 0
#endif

#if USE_ASYNC_LDS
// builtin signature (from hipcc diagnostic): param0 = int4 (vector_size 16) pointer in
// global (__device__/AS1) space, param1 = LDS (AS3) side, then imm offset + cpol.
typedef int v4i_g __attribute__((vector_size(16)));
typedef __attribute__((address_space(1))) v4i_g glb_v4i;
typedef __attribute__((address_space(3))) v4i_g lds_v4i;
static __device__ inline void cp_async_b128(const __bf16* g, __bf16* l) {
    __builtin_amdgcn_global_load_async_to_lds_b128((glb_v4i*)g, (lds_v4i*)l, 0, 0);
}
#endif

static __device__ inline void wait_async0() {
#if __has_builtin(__builtin_amdgcn_s_wait_asynccnt)
    __builtin_amdgcn_s_wait_asynccnt(0);
#else
    asm volatile("s_wait_asynccnt 0x0" ::: "memory");
#endif
}

static __device__ inline __bf16 f2bf(float f) {
    unsigned int u = __builtin_bit_cast(unsigned int, f);
    unsigned int r = u + 0x7FFFu + ((u >> 16) & 1u);   // round-to-nearest-even
    unsigned short s = (unsigned short)(r >> 16);
    union { unsigned short u; __bf16 h; } cv; cv.u = s;
    return cv.h;
}

// ---------------- fp32 -> bf16 convert (weights) ----------------
__global__ __launch_bounds__(256) void cvt_bf16_kernel(const float* __restrict__ in,
                                                       __bf16* __restrict__ out, int n) {
    int i = blockIdx.x * 256 + threadIdx.x;
    if (i < n) out[i] = f2bf(in[i]);
}

// ---------------- offset conv: 3x3, dil=2, pad=2, 256->18 ----------------
__global__ __launch_bounds__(256) void offset_conv_kernel(const float* __restrict__ x,
                                                          const float* __restrict__ w_off,
                                                          const float* __restrict__ b_off,
                                                          float* __restrict__ offs) {
    int idx = blockIdx.x * 256 + threadIdx.x;          // over B*18*HW
    if (idx >= kB * kOffC * kHW) return;
    int hw = idx % kHW;
    int t  = idx / kHW;
    int oc = t % kOffC;
    int b  = t / kOffC;
    int h = hw / kW, w = hw % kW;
    float acc = b_off[oc];
    const float* wp = w_off + (size_t)oc * kC * 9;
    const float* xb = x + (size_t)b * kC * kHW;
    for (int c = 0; c < kC; ++c) {
        const float* xp = xb + (size_t)c * kHW;
        const float* wc = wp + c * 9;
        #pragma unroll
        for (int ky = 0; ky < 3; ++ky) {
            int yy = h + ky * 2 - 2;
            if (yy < 0 || yy >= kH) continue;
            #pragma unroll
            for (int kx = 0; kx < 3; ++kx) {
                int xx = w + kx * 2 - 2;
                if (xx < 0 || xx >= kW) continue;
                acc += xp[yy * kW + xx] * wc[ky * 3 + kx];
            }
        }
    }
    offs[idx] = acc;
}

// ---------------- deformable bilinear im2col: A[p][c*9+k] (bf16) ----------------
static __device__ inline float samp(const float* __restrict__ xc, int y, int xx) {
    bool valid = (y >= 0) & (y < kH) & (xx >= 0) & (xx < kW);
    int yc = min(max(y, 0), kH - 1);
    int xcl = min(max(xx, 0), kW - 1);
    float v = xc[yc * kW + xcl];
    return valid ? v : 0.0f;
}

__global__ __launch_bounds__(256) void deform_im2col_kernel(const float* __restrict__ x,
                                                            const float* __restrict__ offs,
                                                            __bf16* __restrict__ A) {
    const int p = blockIdx.x;                 // 0..M-1
    const int b = p / kHW;
    const int hw = p % kHW;
    const int h = hw / kW, w = hw % kW;
    const float* ob = offs + (size_t)b * kOffC * kHW + hw;
    const float* xb = x + (size_t)b * kC * kHW;
    __bf16* arow = A + (size_t)p * kK;
    for (int t = threadIdx.x; t < kK; t += 256) {
        int c = t / 9, k = t % 9;
        int ky = k / 3, kx = k % 3;
        float dy = ob[(size_t)(2 * k) * kHW];
        float dx = ob[(size_t)(2 * k + 1) * kHW];
        float py = (float)h + (float)(ky * 2 - 2) + dy;
        float px = (float)w + (float)(kx * 2 - 2) + dx;
        float y0f = floorf(py), x0f = floorf(px);
        float wy = py - y0f, wx = px - x0f;
        int y0 = (int)y0f, x0 = (int)x0f;
        const float* xc = xb + (size_t)c * kHW;
        float v00 = samp(xc, y0,     x0);
        float v01 = samp(xc, y0,     x0 + 1);
        float v10 = samp(xc, y0 + 1, x0);
        float v11 = samp(xc, y0 + 1, x0 + 1);
        float val = v00 * (1.f - wy) * (1.f - wx) + v01 * (1.f - wy) * wx
                  + v10 * wy * (1.f - wx)         + v11 * wy * wx;
        arow[t] = f2bf(val);
    }
}

// ---------------- plain 3x3 pad=1 im2col from bf16 NCHW ----------------
__global__ __launch_bounds__(256) void conv_im2col_kernel(const __bf16* __restrict__ in,
                                                          __bf16* __restrict__ A) {
    const int p = blockIdx.x;
    const int b = p / kHW;
    const int hw = p % kHW;
    const int h = hw / kW, w = hw % kW;
    const __bf16* ib = in + (size_t)b * kC * kHW;
    __bf16* arow = A + (size_t)p * kK;
    union { unsigned short u; __bf16 h; } z; z.u = 0;
    for (int t = threadIdx.x; t < kK; t += 256) {
        int c = t / 9, k = t % 9;
        int yy = h + k / 3 - 1;
        int xx = w + k % 3 - 1;
        bool valid = (yy >= 0) & (yy < kH) & (xx >= 0) & (xx < kW);
        arow[t] = valid ? ib[(size_t)c * kHW + yy * kW + xx] : z.h;
    }
}

// ---------------- bf16 WMMA GEMM: C[M][N] = A[M][K] * Bt[N][K]^T ----------------
// block = 256 threads (8 wave32), tile 128(M) x 64(N), BK=32, double-buffered LDS.
// wave grid: 4 (M) x 2 (N); each wave: 32x32 = 2x2 fragments of 16x16.
constexpr int kLdsPad = 40;   // row stride (elems): 80B -> 16 distinct bank groups for frag reads

__global__ __launch_bounds__(256) void gemm_bf16_kernel(const __bf16* __restrict__ A,
                                                        const __bf16* __restrict__ Bt,
                                                        float* __restrict__ Cg,
                                                        int M, int N, int K) {
    __shared__ __bf16 As[2][128 * kLdsPad];
    __shared__ __bf16 Bs[2][64 * kLdsPad];

    const int tid  = threadIdx.x;
    const int m0   = blockIdx.x * 128;
    const int n0   = blockIdx.y * 64;
    const int wv   = tid >> 5;
    const int lane = tid & 31;
    const int wm   = (wv & 3) * 32;       // wave M offset in tile
    const int wn   = (wv >> 2) * 32;      // wave N offset in tile
    const int lrow = lane & 15;
    const int lgrp = lane >> 4;

    v8f acc[2][2];
    #pragma unroll
    for (int i = 0; i < 2; ++i)
        #pragma unroll
        for (int j = 0; j < 2; ++j)
            acc[i][j] = (v8f){0.f, 0.f, 0.f, 0.f, 0.f, 0.f, 0.f, 0.f};

    // global->LDS tile assignments (per thread: 32B of A, 16B of B)
    const int arow = tid >> 1;            // 0..127
    const int acol = (tid & 1) * 16;      // 0 or 16 (elements)
    const int brow = tid >> 2;            // 0..63
    const int bcol = (tid & 3) * 8;       // 0,8,16,24 (elements)

    const __bf16* gA = A  + (size_t)(m0 + arow) * K + acol;
    const __bf16* gB = Bt + (size_t)(n0 + brow) * K + bcol;
    const int aoff = arow * kLdsPad + acol;
    const int boff = brow * kLdsPad + bcol;
    const int nsteps = K / 32;

    // ---- prologue: stage tile 0 into buffer 0 ----
#if USE_ASYNC_LDS
    cp_async_b128(gA,     &As[0][aoff]);
    cp_async_b128(gA + 8, &As[0][aoff + 8]);
    cp_async_b128(gB,     &Bs[0][boff]);
    wait_async0();
    __syncthreads();
#else
    {
        v4u a0 = *reinterpret_cast<const v4u*>(gA);
        v4u a1 = *reinterpret_cast<const v4u*>(gA + 8);
        v4u b0 = *reinterpret_cast<const v4u*>(gB);
        *reinterpret_cast<v4u*>(&As[0][aoff])     = a0;
        *reinterpret_cast<v4u*>(&As[0][aoff + 8]) = a1;
        *reinterpret_cast<v4u*>(&Bs[0][boff])     = b0;
        __syncthreads();
    }
#endif

    for (int i = 0; i < nsteps; ++i) {
        const int p = i & 1;
        const bool has_next = (i + 1) < nsteps;
        const int kn = (i + 1) * 32;

#if USE_ASYNC_LDS
        // stream next tile into the other buffer while this one computes
        if (has_next) {
            cp_async_b128(gA + kn,     &As[1 - p][aoff]);
            cp_async_b128(gA + kn + 8, &As[1 - p][aoff + 8]);
            cp_async_b128(gB + kn,     &Bs[1 - p][boff]);
        }
#else
        v4u na0 = {}, na1 = {}, nb0 = {};
        if (has_next) {
            na0 = *reinterpret_cast<const v4u*>(gA + kn);
            na1 = *reinterpret_cast<const v4u*>(gA + kn + 8);
            nb0 = *reinterpret_cast<const v4u*>(gB + kn);
        }
#endif

        // fragment loads (ISA 7.12.2 layouts) + 4 WMMAs
        FragBF af[2], bfg[2];
        const __bf16* Abase = As[p];
        const __bf16* Bbase = Bs[p];
        #pragma unroll
        for (int mi = 0; mi < 2; ++mi) {
            const __bf16* base = &Abase[(wm + mi * 16 + lrow) * kLdsPad];
            // A: e<8 -> K = lgrp*8+e ; e>=8 -> K = 16+lgrp*8+(e-8)
            af[mi].q[0] = *reinterpret_cast<const v4u*>(base + lgrp * 8);
            af[mi].q[1] = *reinterpret_cast<const v4u*>(base + 16 + lgrp * 8);
        }
        #pragma unroll
        for (int ni = 0; ni < 2; ++ni) {
            const __bf16* base = &Bbase[(wn + ni * 16 + lrow) * kLdsPad];
            // B: e<8 -> K = lgrp*8+e ; e>=8 -> K = 16+lgrp*8+(e-8), N = lane%16 (N-major rows in LDS)
            bfg[ni].q[0] = *reinterpret_cast<const v4u*>(base + lgrp * 8);
            bfg[ni].q[1] = *reinterpret_cast<const v4u*>(base + 16 + lgrp * 8);
        }
        #pragma unroll
        for (int mi = 0; mi < 2; ++mi)
            #pragma unroll
            for (int ni = 0; ni < 2; ++ni)
                acc[mi][ni] = __builtin_amdgcn_wmma_f32_16x16x32_bf16(
                    false, af[mi].v, false, bfg[ni].v,
                    (short)0, acc[mi][ni], false, false);

#if USE_ASYNC_LDS
        if (has_next) {
            wait_async0();       // own async writes landed in LDS
            __syncthreads();     // everyone's landed; also fences reads of buf p
        }
#else
        if (has_next) {
            *reinterpret_cast<v4u*>(&As[1 - p][aoff])     = na0;
            *reinterpret_cast<v4u*>(&As[1 - p][aoff + 8]) = na1;
            *reinterpret_cast<v4u*>(&Bs[1 - p][boff])     = nb0;
            __syncthreads();
        }
#endif
    }

    // store D: VGPR r -> M = r + 8*lanegrp ; N = lane%16
    #pragma unroll
    for (int mi = 0; mi < 2; ++mi)
        #pragma unroll
        for (int ni = 0; ni < 2; ++ni) {
            int col = n0 + wn + ni * 16 + lrow;
            #pragma unroll
            for (int r = 0; r < 8; ++r) {
                int row = m0 + wm + mi * 16 + lgrp * 8 + r;
                Cg[(size_t)row * N + col] = acc[mi][ni][r];
            }
        }
}

// ---------------- deterministic two-pass batchnorm ----------------
__global__ __launch_bounds__(256) void bn_partial_kernel(const float* __restrict__ G,
                                                         float* __restrict__ ps,
                                                         float* __restrict__ ps2) {
    int n = threadIdx.x;                 // channel
    int blk = blockIdx.x;                // 98 blocks x 128 rows
    float s = 0.f, s2 = 0.f;
    for (int r = 0; r < 128; ++r) {
        float v = G[(size_t)(blk * 128 + r) * kN + n];
        s += v; s2 += v * v;
    }
    ps[blk * kN + n] = s;
    ps2[blk * kN + n] = s2;
}

__global__ __launch_bounds__(256) void bn_finalize_kernel(const float* __restrict__ ps,
                                                          const float* __restrict__ ps2,
                                                          const float* __restrict__ gamma,
                                                          const float* __restrict__ beta,
                                                          float* __restrict__ scale,
                                                          float* __restrict__ bias) {
    int n = threadIdx.x;
    float s = 0.f, s2 = 0.f;
    for (int b = 0; b < kM / 128; ++b) { s += ps[b * kN + n]; s2 += ps2[b * kN + n]; }
    float inv = 1.0f / (float)kM;
    float mean = s * inv;
    float var = s2 * inv - mean * mean;
    float sc = gamma[n] * rsqrtf(var + 1e-5f);
    scale[n] = sc;
    bias[n] = beta[n] - mean * sc;
}

// BN1 apply + ReLU, write bf16 NCHW for the next conv
__global__ __launch_bounds__(256) void bn_apply_relu_kernel(const float* __restrict__ G,
                                                            const float* __restrict__ scale,
                                                            const float* __restrict__ bias,
                                                            __bf16* __restrict__ out1) {
    size_t idx = (size_t)blockIdx.x * 256 + threadIdx.x;     // over M*N (row-major N)
    if (idx >= (size_t)kM * kN) return;
    int n = (int)(idx & (kN - 1));
    size_t p = idx >> 8;
    float v = G[idx] * scale[n] + bias[n];
    v = fmaxf(v, 0.f);
    int b = (int)(p / kHW);
    int hw = (int)(p % kHW);
    out1[((size_t)(b * kC + n)) * kHW + hw] = f2bf(v);
}

// BN2 apply + residual + ReLU, NCHW fp32 output
__global__ __launch_bounds__(256) void bn_residual_relu_kernel(const float* __restrict__ G,
                                                               const float* __restrict__ scale,
                                                               const float* __restrict__ bias,
                                                               const float* __restrict__ x,
                                                               float* __restrict__ out) {
    size_t idx = (size_t)blockIdx.x * 256 + threadIdx.x;     // NCHW linear
    if (idx >= (size_t)kB * kC * kHW) return;
    int hw = (int)(idx % kHW);
    size_t t = idx / kHW;
    int c = (int)(t % kC);
    int b = (int)(t / kC);
    size_t p = (size_t)b * kHW + hw;
    float v = G[p * kN + c] * scale[c] + bias[c] + x[idx];
    out[idx] = fmaxf(v, 0.f);
}

// ---------------- host launcher ----------------
extern "C" void kernel_launch(void* const* d_in, const int* in_sizes, int n_in,
                              void* d_out, int out_size, void* d_ws, size_t ws_size,
                              hipStream_t stream) {
    const float* x      = (const float*)d_in[0];
    const float* w_off  = (const float*)d_in[1];
    const float* b_off  = (const float*)d_in[2];
    const float* w1     = (const float*)d_in[3];
    const float* gamma1 = (const float*)d_in[4];
    const float* beta1  = (const float*)d_in[5];
    const float* w2     = (const float*)d_in[6];
    const float* gamma2 = (const float*)d_in[7];
    const float* beta2  = (const float*)d_in[8];
    float* out = (float*)d_out;

    // workspace carve-up (256B aligned)
    char* ws = (char*)d_ws;
    size_t off = 0;
    auto carve = [&](size_t bytes) -> char* {
        char* p = ws + off;
        off = (off + bytes + 255) & ~(size_t)255;
        return p;
    };
    float*  offs  = (float*) carve((size_t)kB * kOffC * kHW * 4);   // 0.90 MB
    __bf16* w1b   = (__bf16*)carve((size_t)kN * kK * 2);            // 1.18 MB
    __bf16* w2b   = (__bf16*)carve((size_t)kN * kK * 2);            // 1.18 MB
    __bf16* Abuf  = (__bf16*)carve((size_t)kM * kK * 2);            // 57.8 MB
    float*  Gbuf  = (float*) carve((size_t)kM * kN * 4);            // 12.8 MB
    __bf16* out1  = (__bf16*)carve((size_t)kM * kN * 2);            // 6.4 MB
    float*  ps    = (float*) carve((size_t)(kM / 128) * kN * 4);
    float*  ps2   = (float*) carve((size_t)(kM / 128) * kN * 4);
    float*  sc1   = (float*) carve(kN * 4);
    float*  bi1   = (float*) carve(kN * 4);
    float*  sc2   = (float*) carve(kN * 4);
    float*  bi2   = (float*) carve(kN * 4);
    (void)ws_size; (void)in_sizes; (void)n_in; (void)out_size;

    const int wcnt = kN * kK;            // 589824 weights per conv

    // 1. weights -> bf16
    cvt_bf16_kernel<<<(wcnt + 255) / 256, 256, 0, stream>>>(w1, w1b, wcnt);
    cvt_bf16_kernel<<<(wcnt + 255) / 256, 256, 0, stream>>>(w2, w2b, wcnt);

    // 2. offset conv (fp32, small)
    offset_conv_kernel<<<(kB * kOffC * kHW + 255) / 256, 256, 0, stream>>>(x, w_off, b_off, offs);

    // 3. deformable bilinear im2col -> Abuf
    deform_im2col_kernel<<<kM, 256, 0, stream>>>(x, offs, Abuf);

    // 4. GEMM1: (M x K) * (K x N) via WMMA bf16
    dim3 ggrid(kM / 128, kN / 64);
    gemm_bf16_kernel<<<ggrid, 256, 0, stream>>>(Abuf, w1b, Gbuf, kM, kN, kK);

    // 5. BN1 + ReLU -> out1 (bf16 NCHW)
    bn_partial_kernel<<<kM / 128, 256, 0, stream>>>(Gbuf, ps, ps2);
    bn_finalize_kernel<<<1, 256, 0, stream>>>(ps, ps2, gamma1, beta1, sc1, bi1);
    bn_apply_relu_kernel<<<(int)(((size_t)kM * kN + 255) / 256), 256, 0, stream>>>(Gbuf, sc1, bi1, out1);

    // 6. im2col for conv2 (reuse Abuf)
    conv_im2col_kernel<<<kM, 256, 0, stream>>>(out1, Abuf);

    // 7. GEMM2 (reuse Gbuf)
    gemm_bf16_kernel<<<ggrid, 256, 0, stream>>>(Abuf, w2b, Gbuf, kM, kN, kK);

    // 8. BN2 + residual + ReLU -> out
    bn_partial_kernel<<<kM / 128, 256, 0, stream>>>(Gbuf, ps, ps2);
    bn_finalize_kernel<<<1, 256, 0, stream>>>(ps, ps2, gamma2, beta2, sc2, bi2);
    bn_residual_relu_kernel<<<(int)(((size_t)kB * kC * kHW + 255) / 256), 256, 0, stream>>>(
        Gbuf, sc2, bi2, x, out);
}
